// GNNModel_6425271075056
// MI455X (gfx1250) — compile-verified, matrix operand order checked by reading, not run on hardware
//
#include <hip/hip_runtime.h>
#include <hip/hip_bf16.h>

// ---------------- types for WMMA ----------------
typedef __bf16 bf16;
typedef bf16  v16bf  __attribute__((ext_vector_type(16)));
typedef bf16  bf16x8 __attribute__((ext_vector_type(8)));
typedef float v8f    __attribute__((ext_vector_type(8)));
typedef float v4f    __attribute__((ext_vector_type(4)));

#define NN 8192
#define HH 64

// ================================================================
// C[N,64] = A[N,N](f32) @ B[N,64](f32), bf16 WMMA with f32 accum.
// Block = 256 threads (8 waves). Each wave: one 16-row M tile x 4 N tiles.
// K-loop unrolled x2 with alternating register sets + LDS buffers:
//   half 0 consumes regs X / Bt[0], prefetches regs Y / Bt[1]
//   half 1 consumes regs Y / Bt[1], prefetches regs X / Bt[0]
// so there is no register rotation and all LDS offsets are immediates.
// ================================================================
__global__ __launch_bounds__(256) void k_gemm_bf16(const float* __restrict__ A,
                                                   const float* __restrict__ B,
                                                   float* __restrict__ C) {
    __shared__ __align__(16) bf16 Bt[2][64][40];   // [buf][n][k], 80B row stride

    const int tid  = threadIdx.x;
    const int lane = tid & 31;
    const int wave = tid >> 5;
    const int lm   = lane & 15;       // row (A) / col (B) within tile
    const int hi   = lane >> 4;       // fragment half select
    const int m0   = blockIdx.x * 128 + wave * 16;

    // cooperative B-load indices: thread -> (n, k-quarter)
    const int bn  = tid & 63;
    const int bkq = tid >> 6;         // 0..3 -> k = bkq*8 .. +7

    v8f acc[4];
#pragma unroll
    for (int t = 0; t < 4; ++t) acc[t] = (v8f)0.0f;

    const float* arow = A + (size_t)(m0 + lm) * NN + 8 * hi;

    // stage B chunk kc into Bt[buf]
    auto stageB = [&](int kc, int buf) {
        const float* bp = B + (size_t)(kc + bkq * 8) * HH + bn;
        bf16x8 pk;
#pragma unroll
        for (int j = 0; j < 8; ++j) pk[j] = (bf16)bp[j * HH];
        *(bf16x8*)&Bt[buf][bn][bkq * 8] = pk;
    };

    // 4 WMMAs from A regs + Bt[buf]
    auto compute = [&](const v4f& a0, const v4f& a1, const v4f& a2, const v4f& a3,
                       int buf) {
        v16bf af;
#pragma unroll
        for (int j = 0; j < 4; ++j) {
            af[j]      = (bf16)a0[j];
            af[4 + j]  = (bf16)a1[j];
            af[8 + j]  = (bf16)a2[j];
            af[12 + j] = (bf16)a3[j];
        }
        const bf16* r0 = &Bt[buf][ 0 + lm][0];
        const bf16* r1 = &Bt[buf][16 + lm][0];
        const bf16* r2 = &Bt[buf][32 + lm][0];
        const bf16* r3 = &Bt[buf][48 + lm][0];
        bf16x8 blo0 = *(const bf16x8*)(r0 + 8 * hi), bhi0 = *(const bf16x8*)(r0 + 16 + 8 * hi);
        bf16x8 blo1 = *(const bf16x8*)(r1 + 8 * hi), bhi1 = *(const bf16x8*)(r1 + 16 + 8 * hi);
        bf16x8 blo2 = *(const bf16x8*)(r2 + 8 * hi), bhi2 = *(const bf16x8*)(r2 + 16 + 8 * hi);
        bf16x8 blo3 = *(const bf16x8*)(r3 + 8 * hi), bhi3 = *(const bf16x8*)(r3 + 16 + 8 * hi);
        v16bf bf0 = __builtin_shufflevector(blo0, bhi0, 0,1,2,3,4,5,6,7,8,9,10,11,12,13,14,15);
        v16bf bf1 = __builtin_shufflevector(blo1, bhi1, 0,1,2,3,4,5,6,7,8,9,10,11,12,13,14,15);
        v16bf bf2 = __builtin_shufflevector(blo2, bhi2, 0,1,2,3,4,5,6,7,8,9,10,11,12,13,14,15);
        v16bf bf3 = __builtin_shufflevector(blo3, bhi3, 0,1,2,3,4,5,6,7,8,9,10,11,12,13,14,15);
        acc[0] = __builtin_amdgcn_wmma_f32_16x16x32_bf16(false, af, false, bf0, (short)0, acc[0], false, false);
        acc[1] = __builtin_amdgcn_wmma_f32_16x16x32_bf16(false, af, false, bf1, (short)0, acc[1], false, false);
        acc[2] = __builtin_amdgcn_wmma_f32_16x16x32_bf16(false, af, false, bf2, (short)0, acc[2], false, false);
        acc[3] = __builtin_amdgcn_wmma_f32_16x16x32_bf16(false, af, false, bf3, (short)0, acc[3], false, false);
    };

    // ---- prologue: stage chunk 0 into X / Bt[0] ----
    v4f xa0, xa1, xa2, xa3, ya0, ya1, ya2, ya3;
    stageB(0, 0);
    xa0 = *(const v4f*)(arow);
    xa1 = *(const v4f*)(arow + 4);
    xa2 = *(const v4f*)(arow + 16);
    xa3 = *(const v4f*)(arow + 20);
    __syncthreads();

    for (int kk = 0; kk < NN; kk += 64) {
        // ---- half 0: consume X / Bt[0]; prefetch kk+32 into Y / Bt[1] ----
        {   // kk+32 is always a valid chunk (NN/32 is even)
            const float* ap = arow + kk + 32;
            ya0 = *(const v4f*)(ap);
            ya1 = *(const v4f*)(ap + 4);
            ya2 = *(const v4f*)(ap + 16);
            ya3 = *(const v4f*)(ap + 20);
            stageB(kk + 32, 1);
        }
        compute(xa0, xa1, xa2, xa3, 0);
        __syncthreads();

        // ---- half 1: consume Y / Bt[1]; prefetch kk+64 into X / Bt[0] ----
        if (kk + 64 < NN) {
            const float* ap = arow + kk + 64;
            xa0 = *(const v4f*)(ap);
            xa1 = *(const v4f*)(ap + 4);
            xa2 = *(const v4f*)(ap + 16);
            xa3 = *(const v4f*)(ap + 20);
            stageB(kk + 64, 0);
        }
        compute(ya0, ya1, ya2, ya3, 1);
        __syncthreads();
    }

    // D layout: VGPR r -> M = r + 8*hi ; lane lm -> N
#pragma unroll
    for (int t = 0; t < 4; ++t) {
        const int n = t * 16 + lm;
#pragma unroll
        for (int r = 0; r < 8; ++r)
            C[(size_t)(m0 + r + 8 * hi) * HH + n] = acc[t][r];
    }
}

// ================================================================
// Out[8192,64] = Xin[8192,K] @ W[K,64]   (tiny dense layers)
// ================================================================
__global__ __launch_bounds__(256) void k_dense(const float* __restrict__ Xin,
                                               const float* __restrict__ W,
                                               float* __restrict__ Out, int K) {
    const int idx = blockIdx.x * 256 + threadIdx.x;
    const int row = idx >> 6, col = idx & 63;
    float s = 0.f;
    for (int k = 0; k < K; ++k) s += Xin[(size_t)row * K + k] * W[k * 64 + col];
    Out[(size_t)row * 64 + col] = s;
}

// ================================================================
// out = LayerNorm(relu(P + bias)) * gamma + beta (+ resid)
// One wave32 per row of 64, two elements per lane, shfl reductions.
// ================================================================
__global__ __launch_bounds__(256) void k_ln(const float* __restrict__ P,
                                            const float* __restrict__ bias,
                                            const float* __restrict__ gamma,
                                            const float* __restrict__ beta,
                                            const float* __restrict__ resid,
                                            float* __restrict__ out) {
    const int lane = threadIdx.x & 31;
    const int wave = threadIdx.x >> 5;
    const int row  = blockIdx.x * 8 + wave;
    const int c    = lane * 2;

    float2 x = *(const float2*)&P[(size_t)row * 64 + c];
    x.x = fmaxf(x.x + bias[c], 0.f);
    x.y = fmaxf(x.y + bias[c + 1], 0.f);

    float s = x.x + x.y;
    for (int off = 16; off > 0; off >>= 1) s += __shfl_xor(s, off, 32);
    const float m  = s * (1.0f / 64.0f);
    const float d0 = x.x - m, d1 = x.y - m;
    float vs = d0 * d0 + d1 * d1;
    for (int off = 16; off > 0; off >>= 1) vs += __shfl_xor(vs, off, 32);
    const float inv = rsqrtf(vs * (1.0f / 64.0f) + 1e-3f);

    float y0 = d0 * inv * gamma[c] + beta[c];
    float y1 = d1 * inv * gamma[c + 1] + beta[c + 1];
    if (resid) {
        y0 += resid[(size_t)row * 64 + c];
        y1 += resid[(size_t)row * 64 + c + 1];
    }
    float2 o; o.x = y0; o.y = y1;
    *(float2*)&out[(size_t)row * 64 + c] = o;
}

// ================================================================
__global__ void k_zero(float* __restrict__ p, int n) {
    const int i = blockIdx.x * 256 + threadIdx.x;
    if (i < n) p[i] = 0.f;
}

// feat*sigmoid(attn) segment-summed per graph (atomics)
__global__ __launch_bounds__(256) void k_pool(const float* __restrict__ h,
                                              const float* __restrict__ Wf,
                                              const float* __restrict__ bfv,
                                              const float* __restrict__ Wa,
                                              const float* __restrict__ bav,
                                              const int* __restrict__ I,
                                              float* __restrict__ g) {
    const int idx = blockIdx.x * 256 + threadIdx.x;
    const int row = idx >> 6, col = idx & 63;
    const float* hr = h + (size_t)row * 64;
    float f = bfv[col], a = bav[col];
    for (int k = 0; k < 64; ++k) {
        const float hv = hr[k];
        f += hv * Wf[k * 64 + col];
        a += hv * Wa[k * 64 + col];
    }
    a = 1.0f / (1.0f + __expf(-a));
    atomicAdd(&g[(size_t)I[row] * 64 + col], f * a);
}

// segment max of z = log1p(relu(signal)); z>=0 so u32-bit atomicMax is ordered
__global__ void k_zmax(const float* __restrict__ X, const int* __restrict__ I,
                       float* __restrict__ zmax) {
    const int i = blockIdx.x * 256 + threadIdx.x;
    if (i >= NN) return;
    const float z = log1pf(fmaxf(X[(size_t)i * 16], 0.f));
    atomicMax((unsigned int*)&zmax[I[i]], __float_as_uint(z));
}

__global__ void k_zexp(const float* __restrict__ X, const int* __restrict__ I,
                       const float* __restrict__ zmax, float* __restrict__ zexp,
                       float* __restrict__ zsum) {
    const int i = blockIdx.x * 256 + threadIdx.x;
    if (i >= NN) return;
    const float z = log1pf(fmaxf(X[(size_t)i * 16], 0.f));
    const float e = __expf(z - zmax[I[i]]);
    zexp[i] = e;
    atomicAdd(&zsum[I[i]], e);
}

__global__ void k_bary(const float* __restrict__ X, const int* __restrict__ I,
                       const float* __restrict__ zexp, const float* __restrict__ zsum,
                       float* __restrict__ bary) {
    const int i = blockIdx.x * 256 + threadIdx.x;
    if (i >= NN) return;
    const int s = I[i];
    const float w = zexp[i] / zsum[s];
    atomicAdd(&bary[s * 3 + 0], w * X[(size_t)i * 16 + 13]);
    atomicAdd(&bary[s * 3 + 1], w * X[(size_t)i * 16 + 14]);
    atomicAdd(&bary[s * 3 + 2], w * X[(size_t)i * 16 + 15]);
}

// out[G,3] = concat(g, bary) @ Wout + bout
__global__ void k_out(const float* __restrict__ g, const float* __restrict__ bary,
                      const float* __restrict__ Wout, const float* __restrict__ bout,
                      float* __restrict__ out) {
    const int idx = blockIdx.x * 256 + threadIdx.x;
    if (idx >= 256 * 3) return;
    const int gi = idx / 3, d = idx % 3;
    float s = bout[d];
    for (int k = 0; k < 64; ++k) s += g[gi * 64 + k] * Wout[k * 3 + d];
    for (int k = 0; k < 3; ++k)  s += bary[gi * 3 + k] * Wout[(64 + k) * 3 + d];
    out[idx] = s;
}

// ================================================================
extern "C" void kernel_launch(void* const* d_in, const int* in_sizes, int n_in,
                              void* d_out, int out_size, void* d_ws, size_t ws_size,
                              hipStream_t stream) {
    const float* X    = (const float*)d_in[0];
    const float* A    = (const float*)d_in[1];
    const int*   I    = (const int*)  d_in[2];
    const float* W0   = (const float*)d_in[3];
    const float* b0   = (const float*)d_in[4];
    const float* g0   = (const float*)d_in[5];
    const float* be0  = (const float*)d_in[6];
    const float* W1   = (const float*)d_in[7];
    const float* b1   = (const float*)d_in[8];
    const float* g1   = (const float*)d_in[9];
    const float* be1  = (const float*)d_in[10];
    const float* Wf   = (const float*)d_in[11];
    const float* bf   = (const float*)d_in[12];
    const float* Wa   = (const float*)d_in[13];
    const float* ba   = (const float*)d_in[14];
    const float* Wout = (const float*)d_in[15];
    const float* bout = (const float*)d_in[16];
    float* out = (float*)d_out;

    float* ws   = (float*)d_ws;
    float* Bbuf = ws;                   // 8192*64
    float* P    = Bbuf + NN * HH;       // 8192*64
    float* h0   = P    + NN * HH;       // 8192*64
    float* h    = h0   + NN * HH;       // 8192*64
    float* g    = h    + NN * HH;       // 256*64   -- contiguous zeroed block:
    float* zmax = g    + 256 * 64;      // 256
    float* zsum = zmax + 256;           // 256
    float* bary = zsum + 256;           // 768      -- end of zeroed block
    float* zexp = bary + 768;           // 8192

    // GCN layer 0
    k_dense<<<NN * HH / 256, 256, 0, stream>>>(X, W0, Bbuf, 16);
    k_gemm_bf16<<<NN / 128, 256, 0, stream>>>(A, Bbuf, P);
    k_ln<<<NN / 8, 256, 0, stream>>>(P, b0, g0, be0, nullptr, h0);
    // GCN layer 1 + residual
    k_dense<<<NN * HH / 256, 256, 0, stream>>>(h0, W1, Bbuf, 64);
    k_gemm_bf16<<<NN / 128, 256, 0, stream>>>(A, Bbuf, P);
    k_ln<<<NN / 8, 256, 0, stream>>>(P, b1, g1, be1, h0, h);
    // pooling heads
    const int nz = 256 * 64 + 256 + 256 + 768;
    k_zero<<<(nz + 255) / 256, 256, 0, stream>>>(g, nz);
    k_pool<<<NN * HH / 256, 256, 0, stream>>>(h, Wf, bf, Wa, ba, I, g);
    k_zmax<<<NN / 256, 256, 0, stream>>>(X, I, zmax);
    k_zexp<<<NN / 256, 256, 0, stream>>>(X, I, zmax, zexp, zsum);
    k_bary<<<NN / 256, 256, 0, stream>>>(X, I, zexp, zsum, bary);
    k_out<<<3, 256, 0, stream>>>(g, bary, Wout, bout, out);
}
